// cmil_mnist_24103356465522
// MI455X (gfx1250) — matrix operand order repaired and checked by reading, not compiled, not run on hardware
//
#include <hip/hip_runtime.h>
#include <math.h>

#define NI 16384
#define NBAG 256

typedef __attribute__((ext_vector_type(8)))  _Float16 v8h;
typedef __attribute__((ext_vector_type(16))) _Float16 v16h;
typedef __attribute__((ext_vector_type(8)))  float    v8f;
typedef __attribute__((ext_vector_type(4)))  unsigned int v4u;
typedef __attribute__((ext_vector_type(8)))  int      v8i;
typedef __attribute__((ext_vector_type(4)))  int      v4i;

__device__ __forceinline__ v16h cat16(v8h a, v8h b) {
  return __builtin_shufflevector(a, b, 0,1,2,3,4,5,6,7,8,9,10,11,12,13,14,15);
}

// ---------------------------------------------------------------------------
// CDNA5 async / TDM codegen probe (NOT launched; count=0 descriptor is an
// architectural NOP per ISA §8.3, so it is safe even if it were executed).
// ---------------------------------------------------------------------------
__global__ void k_cdna5_async_probe(const float* __restrict__ g, float* __restrict__ o)
{
  __shared__ float buf[256];
  unsigned ldsoff = threadIdx.x * 4u;
  const float* p = g + threadIdx.x;
  asm volatile("global_load_async_to_lds_b32 %0, %1, off"
               :: "v"(ldsoff), "v"(p) : "memory");
  asm volatile("s_wait_asynccnt 0x0" ::: "memory");
  v4u g0 = {};
  v8i g1 = {};
  v4i g2 = {};
  v4i g3 = {};
#if defined(__clang_major__) && (__clang_major__ >= 23)
  v8i g4 = {};
  __builtin_amdgcn_tensor_load_to_lds(g0, g1, g2, g3, g4, 0);
#else
  __builtin_amdgcn_tensor_load_to_lds(g0, g1, g2, g3, 0);
#endif
  __builtin_amdgcn_s_wait_tensorcnt((short)0);
  __syncthreads();
  o[threadIdx.x] = buf[threadIdx.x];
}

// ---------------------------------------------------------------------------
// Generic GEMM:  C[M,Nn] = A[M,K](f16,row-major) * W[Nn,K](f16,row-major)^T
// Wave computes a 16 x (16*NT) tile. K multiple of 32, lda == K.
// Software-pipelined: stage k+1 fragments while wmma'ing stage k.
// ---------------------------------------------------------------------------
template<int NT>
__global__ void k_gemm(const _Float16* __restrict__ A, const _Float16* __restrict__ W,
                       _Float16* __restrict__ C16, float* __restrict__ C32,
                       const float* __restrict__ bias,
                       int M, int K, int Nn, int ldc)
{
  const int lane = threadIdx.x & 31;
  const int wave = (int)(((long)blockIdx.x * blockDim.x + threadIdx.x) >> 5);
  const int ngrp = Nn / (16 * NT);
  const int tm = wave / ngrp;
  const int gn = wave - tm * ngrp;
  if (tm * 16 >= M) return;
  const int l16 = lane & 15;
  const int hi  = lane >> 4;

  const _Float16* Ap = A + (long)(tm * 16 + l16) * K + hi * 16;
  const _Float16* Bp[NT];
#pragma unroll
  for (int t = 0; t < NT; ++t)
    Bp[t] = W + (long)((gn * NT + t) * 16 + l16) * K + hi * 16;

  v8f acc[NT];
#pragma unroll
  for (int t = 0; t < NT; ++t) { v8f z = {}; acc[t] = z; }

  // stage 0
  v16h aC = cat16(*(const v8h*)(Ap), *(const v8h*)(Ap + 8));
  v16h bC[NT];
#pragma unroll
  for (int t = 0; t < NT; ++t)
    bC[t] = cat16(*(const v8h*)(Bp[t]), *(const v8h*)(Bp[t] + 8));

#pragma unroll 2
  for (int k0 = 32; k0 < K; k0 += 32) {
    if (k0 + 224 < K) __builtin_prefetch(Ap + k0 + 224, 0, 3);
    v16h aN = cat16(*(const v8h*)(Ap + k0), *(const v8h*)(Ap + k0 + 8));
    v16h bN[NT];
#pragma unroll
    for (int t = 0; t < NT; ++t)
      bN[t] = cat16(*(const v8h*)(Bp[t] + k0), *(const v8h*)(Bp[t] + k0 + 8));
#pragma unroll
    for (int t = 0; t < NT; ++t)
      acc[t] = __builtin_amdgcn_wmma_f32_16x16x32_f16(false, aC, false, bC[t],
                                                      (short)0, acc[t], false, false);
    aC = aN;
#pragma unroll
    for (int t = 0; t < NT; ++t) bC[t] = bN[t];
  }
#pragma unroll
  for (int t = 0; t < NT; ++t)
    acc[t] = __builtin_amdgcn_wmma_f32_16x16x32_f16(false, aC, false, bC[t],
                                                    (short)0, acc[t], false, false);

  const int rbase = tm * 16 + hi * 8;
#pragma unroll
  for (int t = 0; t < NT; ++t) {
    const int col = (gn * NT + t) * 16 + l16;
    const float bv = bias ? bias[col] : 0.0f;
    if (C16) {
#pragma unroll
      for (int r = 0; r < 8; ++r)
        C16[(long)(rbase + r) * ldc + col] = (_Float16)(acc[t][r] + bv);
    }
    if (C32) {
#pragma unroll
      for (int r = 0; r < 8; ++r)
        C32[(long)(rbase + r) * ldc + col] = acc[t][r] + bv;
    }
  }
}

// ---------------------------------------------------------------------------
// conv2 as implicit-im2col WMMA GEMM.  A rows = (n,oy,ox) over NI*49,
// K = 512 (k' = (ky*4+kx)*32 + c, a1 stored NHWC f16 [n][14][14][32]).
// W2h pre-permuted [128][512] to the same k' order.  Nn = 128, NT = 8.
// ---------------------------------------------------------------------------
__global__ void k_gemm_conv2(const _Float16* __restrict__ a1,
                             const _Float16* __restrict__ W2h,
                             _Float16* __restrict__ a2)
{
  const int lane = threadIdx.x & 31;
  const int wave = (int)(((long)blockIdx.x * blockDim.x + threadIdx.x) >> 5);
  const int tilesM = (NI * 49) / 16;
  if (wave >= tilesM) return;
  const int l16 = lane & 15;
  const int hi  = lane >> 4;
  const int row = wave * 16 + l16;
  const int n  = row / 49;
  const int p  = row - n * 49;
  const int oy = p / 7;
  const int ox = p - oy * 7;

  const _Float16* Bp[8];
#pragma unroll
  for (int t = 0; t < 8; ++t) Bp[t] = W2h + (long)(t * 16 + l16) * 512 + hi * 16;

  auto ldA = [&](int b) -> v16h {
    const int ky = b >> 2, kx = b & 3;
    const int iy = 2 * oy - 1 + ky;
    const int ix = 2 * ox - 1 + kx;
    if ((unsigned)iy < 14u && (unsigned)ix < 14u) {
      const _Float16* pa = a1 + (((long)n * 14 + iy) * 14 + ix) * 32 + hi * 16;
      return cat16(*(const v8h*)pa, *(const v8h*)(pa + 8));
    }
    v16h z = {};
    return z;
  };

  v8f acc[8];
#pragma unroll
  for (int t = 0; t < 8; ++t) { v8f z = {}; acc[t] = z; }

  v16h aC = ldA(0);
  v16h bC[8];
#pragma unroll
  for (int t = 0; t < 8; ++t)
    bC[t] = cat16(*(const v8h*)(Bp[t]), *(const v8h*)(Bp[t] + 8));

#pragma unroll 2
  for (int b = 1; b < 16; ++b) {                // b = ky*4+kx == k0/32
    v16h aN = ldA(b);
    v16h bN[8];
#pragma unroll
    for (int t = 0; t < 8; ++t)
      bN[t] = cat16(*(const v8h*)(Bp[t] + b * 32), *(const v8h*)(Bp[t] + b * 32 + 8));
#pragma unroll
    for (int t = 0; t < 8; ++t)
      acc[t] = __builtin_amdgcn_wmma_f32_16x16x32_f16(false, aC, false, bC[t],
                                                      (short)0, acc[t], false, false);
    aC = aN;
#pragma unroll
    for (int t = 0; t < 8; ++t) bC[t] = bN[t];
  }
#pragma unroll
  for (int t = 0; t < 8; ++t)
    acc[t] = __builtin_amdgcn_wmma_f32_16x16x32_f16(false, aC, false, bC[t],
                                                    (short)0, acc[t], false, false);

#pragma unroll
  for (int t = 0; t < 8; ++t) {
    const int col = t * 16 + l16;
#pragma unroll
    for (int r = 0; r < 8; ++r)
      a2[(long)(wave * 16 + hi * 8 + r) * 128 + col] = (_Float16)acc[t][r];
  }
}

// ---------------------------------------------------------------------------
// conv1: 1->32, 4x4, s2, p1 on 28x28 -> NHWC f16 [n][14][14][32]
// ---------------------------------------------------------------------------
__global__ void k_conv1(const float* __restrict__ bag, const float* __restrict__ W1,
                        _Float16* __restrict__ a1)
{
  __shared__ float w[512];
  for (int i = threadIdx.x; i < 512; i += blockDim.x) w[i] = W1[i];
  __syncthreads();
  const long idx = (long)blockIdx.x * blockDim.x + threadIdx.x;
  if (idx >= (long)NI * 196) return;
  const int n = (int)(idx / 196);
  const int p = (int)(idx - (long)n * 196);
  const int oy = p / 14;
  const int ox = p - oy * 14;
  float win[16];
#pragma unroll
  for (int ky = 0; ky < 4; ++ky)
#pragma unroll
    for (int kx = 0; kx < 4; ++kx) {
      const int iy = 2 * oy - 1 + ky, ix = 2 * ox - 1 + kx;
      win[ky * 4 + kx] = ((unsigned)iy < 28u && (unsigned)ix < 28u)
                         ? bag[(long)n * 784 + iy * 28 + ix] : 0.0f;
    }
  _Float16* out = a1 + idx * 32;
#pragma unroll
  for (int c = 0; c < 32; ++c) {
    float s = 0.0f;
#pragma unroll
    for (int k = 0; k < 16; ++k) s += w[c * 16 + k] * win[k];
    out[c] = (_Float16)s;
  }
}

// ---------------------------------------------------------------------------
// BatchNorm helpers
// ---------------------------------------------------------------------------
__global__ void k_stats16(const _Float16* __restrict__ x, long rows, int C,
                          float* __restrict__ st)
{
  const int c = threadIdx.x;              // blockDim.x == C
  float s = 0.0f, q = 0.0f;
  for (long r = blockIdx.x; r < rows; r += gridDim.x) {
    const float v = (float)x[r * C + c];
    s += v; q += v * v;
  }
  atomicAdd(&st[c], s);
  atomicAdd(&st[C + c], q);
}

__global__ void k_stats_fin(float* st, int C, float cnt)
{
  const int c = blockIdx.x * blockDim.x + threadIdx.x;
  if (c >= C) return;
  const float m = st[c] / cnt;
  const float v = st[C + c] / cnt - m * m;
  st[c] = m;
  st[C + c] = rsqrtf(v + 1e-5f);
}

__global__ void k_bn16(_Float16* __restrict__ x, const float* __restrict__ st,
                       long total, int Cm1)
{
  const long i = (long)blockIdx.x * blockDim.x + threadIdx.x;
  if (i >= total) return;
  const int c = (int)(i & Cm1);
  float v = ((float)x[i] - st[c]) * st[Cm1 + 1 + c];
  x[i] = (_Float16)(v > 0.0f ? v : 0.0f);
}

__global__ void k_statsf(const float* __restrict__ x, int Bn, int C, int HW,
                         float* __restrict__ st)
{
  const int c = blockIdx.x;               // one block per channel
  float s = 0.0f, q = 0.0f;
  const int tot = Bn * HW;
  for (int j = threadIdx.x; j < tot; j += blockDim.x) {
    const int b = j / HW, hw = j - b * HW;
    const float v = x[((long)b * C + c) * HW + hw];
    s += v; q += v * v;
  }
  __shared__ float rs[256], rq[256];
  rs[threadIdx.x] = s; rq[threadIdx.x] = q;
  __syncthreads();
  for (int o = 128; o > 0; o >>= 1) {
    if ((int)threadIdx.x < o) { rs[threadIdx.x] += rs[threadIdx.x + o];
                                rq[threadIdx.x] += rq[threadIdx.x + o]; }
    __syncthreads();
  }
  if (threadIdx.x == 0) { st[c] = rs[0]; st[C + c] = rq[0]; }
}

__global__ void k_bnf(float* __restrict__ x, const float* __restrict__ st,
                      long total, int C, int HW)
{
  const long i = (long)blockIdx.x * blockDim.x + threadIdx.x;
  if (i >= total) return;
  const int c = (int)((i / HW) % C);
  const float v = (x[i] - st[c]) * st[C + c];
  x[i] = v > 0.0f ? v : 0.0f;
}

// ---------------------------------------------------------------------------
// Weight preparation (f32 -> permuted f16)
// ---------------------------------------------------------------------------
__global__ void k_prep_w2(const float* __restrict__ W2, _Float16* __restrict__ W2h)
{
  const int i = blockIdx.x * blockDim.x + threadIdx.x;
  if (i >= 128 * 512) return;
  const int o = i / 512, kk = i - o * 512;
  const int b = kk >> 5, c = kk & 31;
  const int ky = b >> 2, kx = b & 3;
  W2h[i] = (_Float16)W2[((o * 32 + c) * 4 + ky) * 4 + kx];
}

__global__ void k_prep_w3(const float* __restrict__ W3, _Float16* __restrict__ W3h)
{
  const int i = blockIdx.x * blockDim.x + threadIdx.x;
  if (i >= 512 * 6272) return;
  const int o = i / 6272, kk = i - o * 6272;
  const int yx = kk >> 7, c = kk & 127;
  const int y = yx / 7, x = yx - y * 7;
  W3h[i] = (_Float16)W3[((o * 128 + c) * 7 + y) * 7 + x];
}

__global__ void k_prep_w4(const float* __restrict__ W4, const float* __restrict__ b4,
                          _Float16* __restrict__ W4h, float* __restrict__ b4f)
{
  const int i = blockIdx.x * blockDim.x + threadIdx.x;
  if (i >= 208 * 512) return;
  const int o = i >> 9, k = i & 511;
  W4h[i] = (o < 200) ? (_Float16)W4[o * 512 + k] : (_Float16)0.0f;
  if (i < 208) b4f[i] = (i < 200) ? b4[i] : 0.0f;
}

__global__ void k_prep_wml(const float* __restrict__ Wm, const float* __restrict__ Wlv,
                           const float* __restrict__ bm, const float* __restrict__ blv,
                           _Float16* __restrict__ Wmlh, float* __restrict__ bmlf)
{
  const int i = blockIdx.x * blockDim.x + threadIdx.x;
  if (i >= 64 * 224) return;
  const int o = i / 224, k = i - o * 224;
  const float* src = (o < 32) ? Wm : Wlv;
  const int oo = o & 31;
  Wmlh[i] = (k < 200) ? (_Float16)src[oo * 200 + k] : (_Float16)0.0f;
  if (i < 64) bmlf[i] = (i < 32) ? bm[i] : blv[i - 32];
}

// ---------------------------------------------------------------------------
// Fills / init
// ---------------------------------------------------------------------------
__global__ void k_zerof(float* p, long n)
{
  for (long i = (long)blockIdx.x * blockDim.x + threadIdx.x; i < n;
       i += (long)gridDim.x * blockDim.x) p[i] = 0.0f;
}
__global__ void k_zero16(_Float16* p, long n)
{
  for (long i = (long)blockIdx.x * blockDim.x + threadIdx.x; i < n;
       i += (long)gridDim.x * blockDim.x) p[i] = (_Float16)0.0f;
}
__global__ void k_init256(unsigned* seg, int* argidx)
{
  const int i = threadIdx.x;
  if (i < 256) { seg[i] = 0u; argidx[i] = 0x7FFFFFFF; }
}

// ---------------------------------------------------------------------------
// zx_q = mu + std*eps ; accumulate KL terms
// ---------------------------------------------------------------------------
__global__ void k_zxq(const float* __restrict__ ml, const float* __restrict__ eps,
                      float* __restrict__ zx, float* __restrict__ acc)
{
  const long i = (long)blockIdx.x * blockDim.x + threadIdx.x;
  float t = 0.0f;
  if (i < (long)NI * 32) {
    const int n = (int)(i >> 5), c = (int)(i & 31);
    const float mu = ml[n * 64 + c];
    const float lv = ml[n * 64 + 32 + c];
    const float sd = expf(0.5f * lv);
    zx[i] = mu + sd * eps[i];
    t = mu * mu + sd * sd - lv - 1.0f;
  }
  __shared__ float rs[256];
  rs[threadIdx.x] = t;
  __syncthreads();
  for (int o = 128; o > 0; o >>= 1) {
    if ((int)threadIdx.x < o) rs[threadIdx.x] += rs[threadIdx.x + o];
    __syncthreads();
  }
  if (threadIdx.x == 0) atomicAdd(&acc[0], rs[0]);
}

__global__ void k_loc(const float* __restrict__ zx, const float* __restrict__ Wfc,
                      const float* __restrict__ bfc, float* __restrict__ loc)
{
  const int n = blockIdx.x * blockDim.x + threadIdx.x;
  if (n >= NI) return;
  float s = bfc[0];
#pragma unroll
  for (int j = 0; j < 32; ++j) s += zx[n * 32 + j] * Wfc[j];
  loc[n] = s;
}

__device__ __forceinline__ unsigned encf(float f)
{
  const int i = __float_as_int(f);
  return (i >= 0) ? ((unsigned)i | 0x80000000u) : ~(unsigned)i;
}

__global__ void k_segmax(const float* __restrict__ loc, const int* __restrict__ bidx,
                         unsigned* __restrict__ seg)
{
  const int n = blockIdx.x * blockDim.x + threadIdx.x;
  if (n >= NI) return;
  atomicMax(&seg[bidx[n]], encf(loc[n]));
}

__global__ void k_argidx(const float* __restrict__ loc, const int* __restrict__ bidx,
                         const unsigned* __restrict__ seg, int* __restrict__ argidx)
{
  const int n = blockIdx.x * blockDim.x + threadIdx.x;
  if (n >= NI) return;
  if (encf(loc[n]) == seg[bidx[n]]) atomicMin(&argidx[bidx[n]], n);
}

// ---------------------------------------------------------------------------
// Decoder (B = 256, fp32)
// ---------------------------------------------------------------------------
__global__ void k_gatherz(const float* __restrict__ zx, const int* __restrict__ argidx,
                          float* __restrict__ z)
{
  const int i = blockIdx.x * blockDim.x + threadIdx.x;
  if (i >= NBAG * 32) return;
  const int b = i >> 5, c = i & 31;
  const int idx = argidx[b];
  z[i] = (idx < NI) ? zx[(long)idx * 32 + c] : 0.0f;
}

__global__ void k_dec1(const float* __restrict__ z, const float* __restrict__ D1,
                       float* __restrict__ o1)
{
  const int i = blockIdx.x * blockDim.x + threadIdx.x;
  if (i >= NBAG * 512) return;
  const int b = i >> 9, o = i & 511;
  float s = 0.0f;
#pragma unroll
  for (int j = 0; j < 32; ++j) s += z[b * 32 + j] * D1[j * 512 + o];
  o1[i] = s;
}

__global__ void k_dec2(const float* __restrict__ o1, const float* __restrict__ D2,
                       float* __restrict__ o2)
{
  const int i = blockIdx.x * blockDim.x + threadIdx.x;
  if (i >= NBAG * 128 * 49) return;
  const int b = i / (128 * 49);
  const int r = i - b * (128 * 49);
  const int o = r / 49;
  const int yx = r - o * 49;
  float s = 0.0f;
  for (int j = 0; j < 512; ++j)
    s += o1[b * 512 + j] * D2[((long)j * 128 + o) * 49 + yx];
  o2[i] = s;
}

__global__ void k_dec3(const float* __restrict__ o2, const float* __restrict__ D3,
                       float* __restrict__ o3)
{
  const int i = blockIdx.x * blockDim.x + threadIdx.x;
  if (i >= NBAG * 32 * 196) return;
  const int b = i / 6272;
  const int r = i - b * 6272;
  const int o = r / 196;
  const int p = r - o * 196;
  const int oy = p / 14, ox = p - (p / 14) * 14;
  int nky = 0, kys[2], iys[2];
#pragma unroll
  for (int ky = 0; ky < 4; ++ky) {
    const int t = oy + 1 - ky;
    if (!(t & 1) && (unsigned)(t >> 1) < 7u) { kys[nky] = ky; iys[nky] = t >> 1; ++nky; }
  }
  int nkx = 0, kxs[2], ixs[2];
#pragma unroll
  for (int kx = 0; kx < 4; ++kx) {
    const int t = ox + 1 - kx;
    if (!(t & 1) && (unsigned)(t >> 1) < 7u) { kxs[nkx] = kx; ixs[nkx] = t >> 1; ++nkx; }
  }
  float s = 0.0f;
  for (int j = 0; j < 128; ++j)
    for (int a = 0; a < nky; ++a)
      for (int c = 0; c < nkx; ++c)
        s += o2[((long)b * 128 + j) * 49 + iys[a] * 7 + ixs[c]] *
             D3[((j * 32 + o) * 4 + kys[a]) * 4 + kxs[c]];
  o3[i] = s;
}

__global__ void k_dec4(const float* __restrict__ o3, const float* __restrict__ D4,
                       const float* __restrict__ d4b, const float* __restrict__ bag,
                       const int* __restrict__ argidx, float* __restrict__ acc)
{
  const int i = blockIdx.x * blockDim.x + threadIdx.x;
  float dd = 0.0f;
  if (i < NBAG * 784) {
    const int b = i / 784;
    const int p = i - b * 784;
    const int oy = p / 28, ox = p - (p / 28) * 28;
    int nky = 0, kys[2], iys[2];
#pragma unroll
    for (int ky = 0; ky < 4; ++ky) {
      const int t = oy + 1 - ky;
      if (!(t & 1) && (unsigned)(t >> 1) < 14u) { kys[nky] = ky; iys[nky] = t >> 1; ++nky; }
    }
    int nkx = 0, kxs[2], ixs[2];
#pragma unroll
    for (int kx = 0; kx < 4; ++kx) {
      const int t = ox + 1 - kx;
      if (!(t & 1) && (unsigned)(t >> 1) < 14u) { kxs[nkx] = kx; ixs[nkx] = t >> 1; ++nkx; }
    }
    float s = d4b[0];
    for (int j = 0; j < 32; ++j)
      for (int a = 0; a < nky; ++a)
        for (int c = 0; c < nkx; ++c)
          s += o3[((long)b * 32 + j) * 196 + iys[a] * 14 + ixs[c]] *
               D4[j * 16 + kys[a] * 4 + kxs[c]];
    const float sig = 1.0f / (1.0f + expf(-s));
    const int idx = argidx[b];
    const float x = (idx < NI) ? bag[(long)idx * 784 + p] : 0.0f;
    const float d = sig - x;
    dd = d * d;
  }
  __shared__ float rs[256];
  rs[threadIdx.x] = dd;
  __syncthreads();
  for (int o = 128; o > 0; o >>= 1) {
    if ((int)threadIdx.x < o) rs[threadIdx.x] += rs[threadIdx.x + o];
    __syncthreads();
  }
  if (threadIdx.x == 0) atomicAdd(&acc[1], rs[0]);
}

__global__ void k_aux(const float* __restrict__ loc, const int* __restrict__ argidx,
                      const float* __restrict__ label, float* __restrict__ acc)
{
  const int b = threadIdx.x;               // single block of 256
  const int idx = argidx[b];
  const float M = (idx < NI) ? loc[idx] : 0.0f;
  const float y = label[b];
  const float term = fmaxf(M, 0.0f) - M * y + log1pf(expf(-fabsf(M)));
  __shared__ float rs[256];
  rs[b] = term;
  __syncthreads();
  for (int o = 128; o > 0; o >>= 1) {
    if (b < o) rs[b] += rs[b + o];
    __syncthreads();
  }
  if (b == 0) acc[2] = rs[0];
}

__global__ void k_final(const float* __restrict__ acc, float* __restrict__ out)
{
  out[0] = acc[1] / (float)(NBAG * 784);                 // recon MSE
  out[1] = 0.5f * acc[0] / (float)((long)NI * 32);       // KL
  out[2] = acc[2] / (float)NBAG;                         // aux BCE
}

// ---------------------------------------------------------------------------
// Host orchestration
// ---------------------------------------------------------------------------
extern "C" void kernel_launch(void* const* d_in, const int* in_sizes, int n_in,
                              void* d_out, int out_size, void* d_ws, size_t ws_size,
                              hipStream_t stream)
{
  (void)in_sizes; (void)n_in; (void)out_size; (void)ws_size;

  const float* bag   = (const float*)d_in[0];
  const int*   bidx  = (const int*)  d_in[1];
  const float* label = (const float*)d_in[2];
  const float* eps   = (const float*)d_in[3];
  const float* W1  = (const float*)d_in[4];
  const float* W2  = (const float*)d_in[6];
  const float* W3  = (const float*)d_in[8];
  const float* W4  = (const float*)d_in[10];
  const float* b4  = (const float*)d_in[11];
  const float* Wm  = (const float*)d_in[12];
  const float* bm  = (const float*)d_in[13];
  const float* Wlv = (const float*)d_in[14];
  const float* blv = (const float*)d_in[15];
  const float* Wfc = (const float*)d_in[16];
  const float* bfc = (const float*)d_in[17];
  const float* D1  = (const float*)d_in[18];
  const float* D2  = (const float*)d_in[20];
  const float* D3  = (const float*)d_in[22];
  const float* D4  = (const float*)d_in[24];
  const float* d4b = (const float*)d_in[25];

  char* ws = (char*)d_ws;
  size_t off = 0;
  auto alloc = [&](size_t bytes) -> size_t {
    size_t o = off; off += (bytes + 255) & ~(size_t)255; return o;
  };
  const size_t o_a1  = alloc((size_t)NI * 196 * 32 * 2);
  const size_t o_a2  = alloc((size_t)NI * 49 * 128 * 2);
  const size_t o_a3  = alloc((size_t)NI * 512 * 2);
  const size_t o_h   = alloc((size_t)NI * 224 * 2);
  const size_t o_ml  = alloc((size_t)NI * 64 * 4);
  const size_t o_zx  = alloc((size_t)NI * 32 * 4);
  const size_t o_loc = alloc((size_t)NI * 4);
  const size_t o_w2  = alloc((size_t)128 * 512 * 2);
  const size_t o_w3  = alloc((size_t)512 * 6272 * 2);
  const size_t o_w4  = alloc((size_t)208 * 512 * 2);
  const size_t o_wml = alloc((size_t)64 * 224 * 2);
  const size_t o_b4  = alloc(208 * 4);
  const size_t o_bml = alloc(64 * 4);
  const size_t o_st  = alloc(1024 * 4);
  const size_t o_seg = alloc(256 * 4);
  const size_t o_arg = alloc(256 * 4);
  const size_t o_z   = alloc((size_t)256 * 32 * 4);
  const size_t o_o1  = alloc((size_t)256 * 512 * 4);
  const size_t o_o2  = alloc((size_t)256 * 128 * 49 * 4);
  const size_t o_o3  = alloc((size_t)256 * 32 * 196 * 4);
  const size_t o_acc = alloc(16 * 4);

  _Float16* a1  = (_Float16*)(ws + o_a1);
  _Float16* a2  = (_Float16*)(ws + o_a2);
  _Float16* a3  = (_Float16*)(ws + o_a3);
  _Float16* hb  = (_Float16*)(ws + o_h);
  float*    ml  = (float*)(ws + o_ml);
  float*    zx  = (float*)(ws + o_zx);
  float*    loc = (float*)(ws + o_loc);
  _Float16* w2h = (_Float16*)(ws + o_w2);
  _Float16* w3h = (_Float16*)(ws + o_w3);
  _Float16* w4h = (_Float16*)(ws + o_w4);
  _Float16* wml = (_Float16*)(ws + o_wml);
  float*    b4f = (float*)(ws + o_b4);
  float*    bml = (float*)(ws + o_bml);
  float*    st  = (float*)(ws + o_st);
  unsigned* seg = (unsigned*)(ws + o_seg);
  int*      arg = (int*)(ws + o_arg);
  float*    zb  = (float*)(ws + o_z);
  float*    o1  = (float*)(ws + o_o1);
  float*    o2  = (float*)(ws + o_o2);
  float*    o3  = (float*)(ws + o_o3);
  float*    acc = (float*)(ws + o_acc);

  auto blk = [](long n) { return (unsigned)((n + 255) / 256); };

  // ---- weight prep + init ----
  k_prep_w2 <<<blk(128 * 512), 256, 0, stream>>>(W2, w2h);
  k_prep_w3 <<<blk((long)512 * 6272), 256, 0, stream>>>(W3, w3h);
  k_prep_w4 <<<blk(208 * 512), 256, 0, stream>>>(W4, b4, w4h, b4f);
  k_prep_wml<<<blk(64 * 224), 256, 0, stream>>>(Wm, Wlv, bm, blv, wml, bml);
  k_zero16  <<<blk((long)NI * 224), 256, 0, stream>>>(hb, (long)NI * 224);
  k_zerof   <<<1, 64, 0, stream>>>(acc, 16);
  k_init256 <<<1, 256, 0, stream>>>(seg, arg);

  // ---- encoder ----
  k_conv1<<<blk((long)NI * 196), 256, 0, stream>>>(bag, W1, a1);
  k_zerof<<<4, 256, 0, stream>>>(st, 1024);
  k_stats16<<<2048, 32, 0, stream>>>(a1, (long)NI * 196, 32, st);
  k_stats_fin<<<1, 32, 0, stream>>>(st, 32, (float)((long)NI * 196));
  k_bn16<<<blk((long)NI * 196 * 32), 256, 0, stream>>>(a1, st, (long)NI * 196 * 32, 31);

  k_gemm_conv2<<<(NI * 49 / 16) * 32 / 256, 256, 0, stream>>>(a1, w2h, a2);
  k_zerof<<<4, 256, 0, stream>>>(st, 1024);
  k_stats16<<<1024, 128, 0, stream>>>(a2, (long)NI * 49, 128, st);
  k_stats_fin<<<1, 128, 0, stream>>>(st, 128, (float)((long)NI * 49));
  k_bn16<<<blk((long)NI * 49 * 128), 256, 0, stream>>>(a2, st, (long)NI * 49 * 128, 127);

  // conv3: [NI,6272] x [512,6272]^T
  k_gemm<8><<<(NI / 16) * (512 / 128) * 32 / 256, 256, 0, stream>>>(
      a2, w3h, a3, nullptr, nullptr, NI, 6272, 512, 512);
  k_zerof<<<4, 256, 0, stream>>>(st, 1024);
  k_stats16<<<512, 512, 0, stream>>>(a3, (long)NI, 512, st);
  k_stats_fin<<<1, 512, 0, stream>>>(st, 512, (float)NI);
  k_bn16<<<blk((long)NI * 512), 256, 0, stream>>>(a3, st, (long)NI * 512, 511);

  // conv4: [NI,512] x [208,512]^T  (+b4) -> h [NI,224] (padded)
  k_gemm<1><<<(NI / 16) * 13 * 32 / 256, 256, 0, stream>>>(
      a3, w4h, hb, nullptr, b4f, NI, 512, 208, 224);
  // mu|logvar: [NI,224] x [64,224]^T -> ml f32
  k_gemm<4><<<(NI / 16) * 1 * 32 / 256, 256, 0, stream>>>(
      hb, wml, nullptr, ml, bml, NI, 224, 64, 64);

  // ---- reparameterize + KL ----
  k_zxq<<<blk((long)NI * 32), 256, 0, stream>>>(ml, eps, zx, acc);
  k_loc<<<blk(NI), 256, 0, stream>>>(zx, Wfc, bfc, loc);
  k_segmax<<<blk(NI), 256, 0, stream>>>(loc, bidx, seg);
  k_argidx<<<blk(NI), 256, 0, stream>>>(loc, bidx, seg, arg);

  // ---- decoder (B=256) ----
  k_gatherz<<<blk(NBAG * 32), 256, 0, stream>>>(zx, arg, zb);
  k_dec1<<<blk(NBAG * 512), 256, 0, stream>>>(zb, D1, o1);
  k_zerof<<<4, 256, 0, stream>>>(st, 1024);
  k_statsf<<<512, 256, 0, stream>>>(o1, NBAG, 512, 1, st);
  k_stats_fin<<<1, 512, 0, stream>>>(st, 512, (float)NBAG);
  k_bnf<<<blk((long)NBAG * 512), 256, 0, stream>>>(o1, st, (long)NBAG * 512, 512, 1);

  k_dec2<<<blk((long)NBAG * 128 * 49), 256, 0, stream>>>(o1, D2, o2);
  k_zerof<<<4, 256, 0, stream>>>(st, 1024);
  k_statsf<<<128, 256, 0, stream>>>(o2, NBAG, 128, 49, st);
  k_stats_fin<<<1, 128, 0, stream>>>(st, 128, (float)(NBAG * 49));
  k_bnf<<<blk((long)NBAG * 128 * 49), 256, 0, stream>>>(o2, st, (long)NBAG * 128 * 49, 128, 49);

  k_dec3<<<blk((long)NBAG * 32 * 196), 256, 0, stream>>>(o2, D3, o3);
  k_zerof<<<4, 256, 0, stream>>>(st, 1024);
  k_statsf<<<32, 256, 0, stream>>>(o3, NBAG, 32, 196, st);
  k_stats_fin<<<1, 32, 0, stream>>>(st, 32, (float)(NBAG * 196));
  k_bnf<<<blk((long)NBAG * 32 * 196), 256, 0, stream>>>(o3, st, (long)NBAG * 32 * 196, 32, 196);

  k_dec4<<<blk((long)NBAG * 784), 256, 0, stream>>>(o3, D4, d4b, bag, arg, acc);
  k_aux<<<1, 256, 0, stream>>>(loc, arg, label, acc);
  k_final<<<1, 1, 0, stream>>>(acc, (float*)d_out);
}
